// main_model_16664473109114
// MI455X (gfx1250) — compile-verified, matrix operand order checked by reading, not run on hardware
//
#include <hip/hip_runtime.h>
#include <cstddef>

// ---------------------------------------------------------------------------
// Model dims
// ---------------------------------------------------------------------------
#define BATCH     4
#define SEQ       256
#define FEAT      80
#define DMODEL    2048
#define ED        4096
#define NSTATE    16
#define DT_RANK   2
#define KCONV     4
#define NLAYERS   6
#define NROWS     (BATCH * SEQ)
#define DBC_N     (DT_RANK + 2 * NSTATE)   // 34

typedef __attribute__((ext_vector_type(8)))  __bf16 v8bf;
typedef __attribute__((ext_vector_type(16))) __bf16 v16bf;
typedef __attribute__((ext_vector_type(8)))  float  v8f;
typedef int v4i_gcc __attribute__((vector_size(16)));   // matches builtin param

#define ACT_NONE  0
#define ACT_SILU  1
#define ACT_LRELU 2

// ---------------------------------------------------------------------------
// CDNA5 async global->LDS copy (ASYNCcnt-tracked), with safe fallback
// ---------------------------------------------------------------------------
#if defined(__has_builtin)
#  if __has_builtin(__builtin_amdgcn_global_load_async_to_lds_b128)
#    define HAVE_ASYNC_LDS 1
#  endif
#  if __has_builtin(__builtin_amdgcn_s_wait_asynccnt)
#    define HAVE_WAIT_ASYNC 1
#  endif
#endif

__device__ __forceinline__ void copy16_g2l(const __bf16* g, __bf16* l)
{
#if HAVE_ASYNC_LDS
    __builtin_amdgcn_global_load_async_to_lds_b128(
        (__attribute__((address_space(1))) v4i_gcc*)g,
        (__attribute__((address_space(3))) v4i_gcc*)l, 0, 0);
#else
    *reinterpret_cast<uint4*>(l) = *reinterpret_cast<const uint4*>(g);
#endif
}

// Double-buffer partial wait: the 5 just-issued next-tile copies may remain
// in flight; everything older (the current tile) is guaranteed complete.
__device__ __forceinline__ void wait_async_prev()
{
#if HAVE_ASYNC_LDS
#  if HAVE_WAIT_ASYNC
    __builtin_amdgcn_s_wait_asynccnt(5);
#  else
    asm volatile("s_wait_asynccnt 0x5" ::: "memory");
#  endif
#endif
}

__device__ __forceinline__ void wait_async_all()
{
#if HAVE_ASYNC_LDS
#  if HAVE_WAIT_ASYNC
    __builtin_amdgcn_s_wait_asynccnt(0);
#  else
    asm volatile("s_wait_asynccnt 0x0" ::: "memory");
#  endif
#endif
}

// ---------------------------------------------------------------------------
// GEMM epilogue helper
// ---------------------------------------------------------------------------
__device__ __forceinline__ void epilogue_store(
    float v, const float* bias, const float* residual,
    void* Cv, int out_bf16, long long ldc, int gr, int gc, int act)
{
    if (bias)     v += bias[gc];
    if (residual) v += residual[(size_t)gr * ldc + gc];
    if (act == ACT_SILU)       v = v / (1.0f + __expf(-v));
    else if (act == ACT_LRELU) v = (v > 0.0f) ? v : 0.01f * v;
    if (out_bf16) ((__bf16*)Cv)[(size_t)gr * ldc + gc] = (__bf16)v;
    else          ((float*)Cv)[(size_t)gr * ldc + gc] = v;
}

// ---------------------------------------------------------------------------
// Fast GEMM: 64(M) x 256(N) x 32(K) block tile, double-buffered async LDS.
// 8 waves (2x4); each wave computes 2x4 WMMA 16x16x32 fragments (32x64).
// Requires M%64==0, N%256==0, K%32==0.  A: bf16 [M][K]; Bt: bf16 [N][K].
// ---------------------------------------------------------------------------
#define BM  64
#define BN  128      // generic-kernel N tile
#define FBN 256      // fast-kernel N tile
#define BK  32
#define TS  40       // LDS row stride in bf16 (80B: 16B-aligned, bank-spread)
#define ASZ (BM * TS)
#define BSZ (FBN * TS)

__global__ __launch_bounds__(256)
void wmma_gemm_fast(const __bf16* __restrict__ A, const __bf16* __restrict__ Bt,
                    const float* __restrict__ bias,
                    const float* __restrict__ residual,
                    void* __restrict__ Cv, int out_bf16, long long ldc,
                    int K, int N, int act)
{
    __shared__ alignas(16) __bf16 As[2 * ASZ];
    __shared__ alignas(16) __bf16 Bs[2 * BSZ];

    const int tileM = blockIdx.y * BM;
    const int tileN = blockIdx.x * FBN;

    const int tid    = threadIdx.x;
    const int lane   = tid & 31;
    const int wid    = tid >> 5;
    const int waveM  = wid & 1;       // 2 waves along M (32 rows each)
    const int waveN  = wid >> 1;      // 4 waves along N (64 cols each)
    const int lane16 = lane & 15;
    const int kh     = (lane >> 4) * 8;

    // staging: each thread owns one 16B segment of A and four of B per K-step
    const int r4 = tid >> 2;          // 0..63
    const int s8 = (tid & 3) * 8;     // bf16 offset of the 16B segment
    const __bf16* gA = A + (size_t)(tileM + r4) * K + s8;
    const __bf16* gB0 = Bt + (size_t)(tileN +   0 + r4) * K + s8;
    const __bf16* gB1 = Bt + (size_t)(tileN +  64 + r4) * K + s8;
    const __bf16* gB2 = Bt + (size_t)(tileN + 128 + r4) * K + s8;
    const __bf16* gB3 = Bt + (size_t)(tileN + 192 + r4) * K + s8;
    const int lA  = r4 * TS + s8;
    const int lB0 = (  0 + r4) * TS + s8;
    const int lB1 = ( 64 + r4) * TS + s8;
    const int lB2 = (128 + r4) * TS + s8;
    const int lB3 = (192 + r4) * TS + s8;

    v8f zero = {};
    v8f acc[2][4];
    #pragma unroll
    for (int fm = 0; fm < 2; ++fm)
        #pragma unroll
        for (int fn = 0; fn < 4; ++fn) acc[fm][fn] = zero;

    // prologue: stage tile 0 into buffer 0
    copy16_g2l(gA,  &As[lA]);
    copy16_g2l(gB0, &Bs[lB0]);
    copy16_g2l(gB1, &Bs[lB1]);
    copy16_g2l(gB2, &Bs[lB2]);
    copy16_g2l(gB3, &Bs[lB3]);
    gA += BK; gB0 += BK; gB1 += BK; gB2 += BK; gB3 += BK;

    int cur = 0;
    // steady state: K/BK - 1 iterations, unconditional issue + partial wait
    for (int it = 0; it < K / BK - 1; ++it) {
        const int nxt = cur ^ 1;
        copy16_g2l(gA,  &As[nxt * ASZ + lA]);
        copy16_g2l(gB0, &Bs[nxt * BSZ + lB0]);
        copy16_g2l(gB1, &Bs[nxt * BSZ + lB1]);
        copy16_g2l(gB2, &Bs[nxt * BSZ + lB2]);
        copy16_g2l(gB3, &Bs[nxt * BSZ + lB3]);
        gA += BK; gB0 += BK; gB1 += BK; gB2 += BK; gB3 += BK;
        wait_async_prev();
        __syncthreads();

        const __bf16* Ab = &As[cur * ASZ];
        const __bf16* Bb = &Bs[cur * BSZ];
        #pragma unroll
        for (int fm = 0; fm < 2; ++fm) {
            const int rowA = waveM * 32 + fm * 16 + lane16;
            v8bf alo = *reinterpret_cast<const v8bf*>(&Ab[rowA * TS + kh]);
            v8bf ahi = *reinterpret_cast<const v8bf*>(&Ab[rowA * TS + kh + 16]);
            v16bf af = __builtin_shufflevector(alo, ahi,
                0,1,2,3,4,5,6,7,8,9,10,11,12,13,14,15);
            #pragma unroll
            for (int fn = 0; fn < 4; ++fn) {
                const int colB = waveN * 64 + fn * 16 + lane16;
                v8bf blo = *reinterpret_cast<const v8bf*>(&Bb[colB * TS + kh]);
                v8bf bhi = *reinterpret_cast<const v8bf*>(&Bb[colB * TS + kh + 16]);
                v16bf bf = __builtin_shufflevector(blo, bhi,
                    0,1,2,3,4,5,6,7,8,9,10,11,12,13,14,15);
                acc[fm][fn] = __builtin_amdgcn_wmma_f32_16x16x32_bf16(
                    false, af, false, bf, (short)0, acc[fm][fn], false, false);
            }
        }
        __syncthreads();   // all waves done reading `cur` before its re-fill
        cur = nxt;
    }

    // drain: last tile
    wait_async_all();
    __syncthreads();
    {
        const __bf16* Ab = &As[cur * ASZ];
        const __bf16* Bb = &Bs[cur * BSZ];
        #pragma unroll
        for (int fm = 0; fm < 2; ++fm) {
            const int rowA = waveM * 32 + fm * 16 + lane16;
            v8bf alo = *reinterpret_cast<const v8bf*>(&Ab[rowA * TS + kh]);
            v8bf ahi = *reinterpret_cast<const v8bf*>(&Ab[rowA * TS + kh + 16]);
            v16bf af = __builtin_shufflevector(alo, ahi,
                0,1,2,3,4,5,6,7,8,9,10,11,12,13,14,15);
            #pragma unroll
            for (int fn = 0; fn < 4; ++fn) {
                const int colB = waveN * 64 + fn * 16 + lane16;
                v8bf blo = *reinterpret_cast<const v8bf*>(&Bb[colB * TS + kh]);
                v8bf bhi = *reinterpret_cast<const v8bf*>(&Bb[colB * TS + kh + 16]);
                v16bf bf = __builtin_shufflevector(blo, bhi,
                    0,1,2,3,4,5,6,7,8,9,10,11,12,13,14,15);
                acc[fm][fn] = __builtin_amdgcn_wmma_f32_16x16x32_bf16(
                    false, af, false, bf, (short)0, acc[fm][fn], false, false);
            }
        }
    }

    const int hiM = (lane >> 4) * 8;
    #pragma unroll
    for (int fm = 0; fm < 2; ++fm)
        #pragma unroll
        for (int fn = 0; fn < 4; ++fn)
            #pragma unroll
            for (int r = 0; r < 8; ++r)
                epilogue_store(acc[fm][fn][r], bias, residual, Cv, out_bf16, ldc,
                               tileM + waveM * 32 + fm * 16 + hiM + r,
                               tileN + waveN * 64 + fn * 16 + lane16, act);
}

// ---- generic guarded path (ragged M/N/K; only small GEMMs use this) -------
__global__ __launch_bounds__(256)
void wmma_gemm_generic(const __bf16* __restrict__ A, const __bf16* __restrict__ Bt,
                       const float* __restrict__ bias,
                       const float* __restrict__ residual,
                       void* __restrict__ Cv, int out_bf16, long long ldc,
                       int M, int N, int K, int act)
{
    __shared__ alignas(16) __bf16 Asg[BM * TS];
    __shared__ alignas(16) __bf16 Bsg[BN * TS];

    const int tileM = blockIdx.y * BM;
    const int tileN = blockIdx.x * BN;
    const int tid    = threadIdx.x;
    const int lane   = tid & 31;
    const int wid    = tid >> 5;
    const int waveM  = wid & 1;
    const int waveN  = wid >> 1;
    const int lane16 = lane & 15;
    const int kh     = (lane >> 4) * 8;

    v8f zero = {};
    v8f acc[2][2] = { { zero, zero }, { zero, zero } };

    for (int k0 = 0; k0 < K; k0 += BK) {
        #pragma unroll
        for (int i = tid; i < BM * BK; i += 256) {
            int r = i >> 5, c = i & 31;
            int gr = tileM + r, gk = k0 + c;
            Asg[r * TS + c] = (gr < M && gk < K) ? A[(size_t)gr * K + gk]
                                                 : (__bf16)0.0f;
        }
        #pragma unroll
        for (int i = tid; i < BN * BK; i += 256) {
            int n = i >> 5, c = i & 31;
            int gn = tileN + n, gk = k0 + c;
            Bsg[n * TS + c] = (gn < N && gk < K) ? Bt[(size_t)gn * K + gk]
                                                 : (__bf16)0.0f;
        }
        __syncthreads();

        #pragma unroll
        for (int fm = 0; fm < 2; ++fm) {
            const int rowA = waveM * 32 + fm * 16 + lane16;
            v8bf alo = *reinterpret_cast<const v8bf*>(&Asg[rowA * TS + kh]);
            v8bf ahi = *reinterpret_cast<const v8bf*>(&Asg[rowA * TS + kh + 16]);
            v16bf af = __builtin_shufflevector(alo, ahi,
                0,1,2,3,4,5,6,7,8,9,10,11,12,13,14,15);
            #pragma unroll
            for (int fn = 0; fn < 2; ++fn) {
                const int colB = waveN * 32 + fn * 16 + lane16;
                v8bf blo = *reinterpret_cast<const v8bf*>(&Bsg[colB * TS + kh]);
                v8bf bhi = *reinterpret_cast<const v8bf*>(&Bsg[colB * TS + kh + 16]);
                v16bf bf = __builtin_shufflevector(blo, bhi,
                    0,1,2,3,4,5,6,7,8,9,10,11,12,13,14,15);
                acc[fm][fn] = __builtin_amdgcn_wmma_f32_16x16x32_bf16(
                    false, af, false, bf, (short)0, acc[fm][fn], false, false);
            }
        }
        __syncthreads();
    }

    const int hiM = (lane >> 4) * 8;
    #pragma unroll
    for (int fm = 0; fm < 2; ++fm)
        #pragma unroll
        for (int fn = 0; fn < 2; ++fn)
            #pragma unroll
            for (int r = 0; r < 8; ++r) {
                int gr = tileM + waveM * 32 + fm * 16 + hiM + r;
                int gc = tileN + waveN * 32 + fn * 16 + lane16;
                if (gr < M && gc < N)
                    epilogue_store(acc[fm][fn][r], bias, residual, Cv, out_bf16,
                                   ldc, gr, gc, act);
            }
}

// ---------------------------------------------------------------------------
// Tiled cast+transpose: W fp32 [K][N] -> Wt bf16 [N][K]
// ---------------------------------------------------------------------------
__global__ __launch_bounds__(256)
void castT_kernel(const float* __restrict__ in, __bf16* __restrict__ out,
                  int K, int N)
{
    __shared__ float t[32][33];
    const int n0 = blockIdx.x * 32;
    const int k0 = blockIdx.y * 32;
    const int tx = threadIdx.x;        // 0..31
    const int ty = threadIdx.y;        // 0..7
    #pragma unroll
    for (int j = 0; j < 4; ++j) {
        int k = k0 + ty + j * 8, n = n0 + tx;
        t[ty + j * 8][tx] = (k < K && n < N) ? in[(size_t)k * N + n] : 0.0f;
    }
    __syncthreads();
    #pragma unroll
    for (int j = 0; j < 4; ++j) {
        int n = n0 + ty + j * 8, k = k0 + tx;
        if (n < N && k < K)
            out[(size_t)n * K + k] = (__bf16)t[tx][ty + j * 8];
    }
}

// ---------------------------------------------------------------------------
// x (B,F,L) -> xT bf16 (B*L, F)
// ---------------------------------------------------------------------------
__global__ __launch_bounds__(256)
void transpose_x_kernel(const float* __restrict__ x, __bf16* __restrict__ xT)
{
    int idx = blockIdx.x * 256 + threadIdx.x;
    if (idx < BATCH * SEQ * FEAT) {
        int f  = idx % FEAT;
        int bl = idx / FEAT;
        int l  = bl % SEQ;
        int b  = bl / SEQ;
        xT[idx] = (__bf16)x[((size_t)b * FEAT + f) * SEQ + l];
    }
}

// ---------------------------------------------------------------------------
// RMSNorm over DMODEL (bf16 output for the GEMM)
// ---------------------------------------------------------------------------
__global__ __launch_bounds__(256)
void rmsnorm_kernel(const float* __restrict__ h, const float* __restrict__ w,
                    __bf16* __restrict__ out)
{
    __shared__ float red[256];
    const int row = blockIdx.x;
    const float* hr = h + (size_t)row * DMODEL;
    float s = 0.0f;
    for (int i = threadIdx.x; i < DMODEL; i += 256) { float v = hr[i]; s += v * v; }
    red[threadIdx.x] = s;
    __syncthreads();
    for (int off = 128; off > 0; off >>= 1) {
        if (threadIdx.x < off) red[threadIdx.x] += red[threadIdx.x + off];
        __syncthreads();
    }
    const float scale = rsqrtf(red[0] / (float)DMODEL + 1e-5f);
    for (int i = threadIdx.x; i < DMODEL; i += 256)
        out[(size_t)row * DMODEL + i] = (__bf16)(hr[i] * scale * w[i]);
}

// ---------------------------------------------------------------------------
// Depthwise causal conv (K=4) + SiLU; fp32 (scan) and bf16 (GEMM) outputs
// ---------------------------------------------------------------------------
__global__ __launch_bounds__(256)
void conv_silu_kernel(const float* __restrict__ xz, const float* __restrict__ cw,
                      const float* __restrict__ cb,
                      float* __restrict__ xa, __bf16* __restrict__ xa16)
{
    size_t idx = (size_t)blockIdx.x * 256 + threadIdx.x;   // B*L*ED
    int e  = (int)(idx % ED);
    size_t bl = idx / ED;
    int l  = (int)(bl % SEQ);
    int b  = (int)(bl / SEQ);
    float acc = cb[e];
    #pragma unroll
    for (int k = 0; k < KCONV; ++k) {
        int ls = l - (KCONV - 1) + k;
        if (ls >= 0)
            acc += cw[(size_t)e * KCONV + k] *
                   xz[((size_t)b * SEQ + ls) * (2 * ED) + e];
    }
    float v = acc / (1.0f + __expf(-acc));
    xa[idx]   = v;
    xa16[idx] = (__bf16)v;
}

// ---------------------------------------------------------------------------
// dt = softplus(dbc[:, :2] @ W_dt + b_dt)
// ---------------------------------------------------------------------------
__global__ __launch_bounds__(256)
void dt_kernel(const float* __restrict__ dbc, const float* __restrict__ wdt,
               const float* __restrict__ bdt, float* __restrict__ dtb)
{
    size_t idx = (size_t)blockIdx.x * 256 + threadIdx.x;   // B*L*ED
    int e = (int)(idx % ED);
    size_t row = idx / ED;
    float v = dbc[row * DBC_N + 0] * wdt[e]
            + dbc[row * DBC_N + 1] * wdt[ED + e]
            + bdt[e];
    dtb[idx] = (v > 20.0f) ? v : log1pf(__expf(v));
}

// ---------------------------------------------------------------------------
// Selective scan (fp32 state math), bf16 gated output for W_out GEMM
// ---------------------------------------------------------------------------
__global__ __launch_bounds__(256)
void scan_kernel(const float* __restrict__ dtb, const float* __restrict__ xa,
                 const float* __restrict__ dbc, const float* __restrict__ xz,
                 const float* __restrict__ Alog, const float* __restrict__ Dp,
                 __bf16* __restrict__ yb)
{
    int c = blockIdx.x * 256 + threadIdx.x;   // B*ED channels
    int e = c % ED;
    int b = c / ED;

    float Ac[NSTATE];
    #pragma unroll
    for (int n = 0; n < NSTATE; ++n)
        Ac[n] = -__expf(Alog[(size_t)e * NSTATE + n]);
    const float dpe = Dp[e];

    float h[NSTATE];
    #pragma unroll
    for (int n = 0; n < NSTATE; ++n) h[n] = 0.0f;

    for (int l = 0; l < SEQ; ++l) {
        size_t row = (size_t)b * SEQ + l;
        float dt_t = dtb[row * ED + e];
        float xa_t = xa[row * ED + e];
        const float* dr = dbc + row * DBC_N;
        float dx = dt_t * xa_t;
        float y = 0.0f;
        #pragma unroll
        for (int n = 0; n < NSTATE; ++n) {
            float hn = __expf(dt_t * Ac[n]) * h[n] + dx * dr[DT_RANK + n];
            h[n] = hn;
            y += hn * dr[DT_RANK + NSTATE + n];
        }
        float z = xz[row * (2 * ED) + ED + e];
        yb[row * ED + e] = (__bf16)((y + dpe * xa_t) * (z / (1.0f + __expf(-z))));
    }
}

// ---------------------------------------------------------------------------
// extract last-token rows of h -> bf16 [BATCH][DMODEL]
// ---------------------------------------------------------------------------
__global__ __launch_bounds__(256)
void last_rows_cast_kernel(const float* __restrict__ h, __bf16* __restrict__ out)
{
    int i = blockIdx.x * 256 + threadIdx.x;     // BATCH*DMODEL
    if (i < BATCH * DMODEL) {
        int b = i / DMODEL, c = i % DMODEL;
        out[i] = (__bf16)h[((size_t)b * SEQ + SEQ - 1) * DMODEL + c];
    }
}

// ---------------------------------------------------------------------------
// out[b] = sigmoid(g2[b,:] @ W_h3 + b_h3)
// ---------------------------------------------------------------------------
__global__ void head_out_kernel(const __bf16* __restrict__ g2,
                                const float* __restrict__ w,
                                const float* __restrict__ bb,
                                float* __restrict__ out)
{
    int b = threadIdx.x;
    if (b < BATCH) {
        float acc = bb[0];
        for (int k = 0; k < 256; ++k) acc += (float)g2[b * 256 + k] * w[k];
        out[b] = 1.0f / (1.0f + __expf(-acc));
    }
}

// ---------------------------------------------------------------------------
// Host side
// ---------------------------------------------------------------------------
static inline void launch_gemm(const __bf16* A, const __bf16* Bt,
                               const float* bias, const float* residual,
                               void* C, int out_bf16, long long ldc,
                               int M, int N, int K, int act, hipStream_t s)
{
    if ((M % BM) == 0 && (N % FBN) == 0 && (K % BK) == 0 && K >= 2 * BK) {
        dim3 grid(N / FBN, M / BM);
        wmma_gemm_fast<<<grid, 256, 0, s>>>(A, Bt, bias, residual, C, out_bf16,
                                            ldc, K, N, act);
    } else {
        dim3 grid((N + BN - 1) / BN, (M + BM - 1) / BM);
        wmma_gemm_generic<<<grid, 256, 0, s>>>(A, Bt, bias, residual, C, out_bf16,
                                               ldc, M, N, K, act);
    }
}

static inline void launch_castT(const float* W, __bf16* Wt, int K, int N,
                                hipStream_t s)
{
    dim3 grid((N + 31) / 32, (K + 31) / 32);
    castT_kernel<<<grid, dim3(32, 8), 0, s>>>(W, Wt, K, N);
}

extern "C" void kernel_launch(void* const* d_in, const int* in_sizes, int n_in,
                              void* d_out, int out_size, void* d_ws, size_t ws_size,
                              hipStream_t stream)
{
    const float* x      = (const float*)d_in[0];
    const float* W_emb  = (const float*)d_in[1];
    const float* b_emb  = (const float*)d_in[2];
    const float* norm_w = (const float*)d_in[3];
    const float* W_in   = (const float*)d_in[4];
    const float* b_in   = (const float*)d_in[5];
    const float* conv_w = (const float*)d_in[6];
    const float* conv_b = (const float*)d_in[7];
    const float* W_x    = (const float*)d_in[8];
    const float* W_dt   = (const float*)d_in[9];
    const float* b_dt   = (const float*)d_in[10];
    const float* A_log  = (const float*)d_in[11];
    const float* Dp     = (const float*)d_in[12];
    const float* W_out  = (const float*)d_in[13];
    const float* b_out  = (const float*)d_in[14];
    const float* W_f    = (const float*)d_in[15];
    const float* b_f    = (const float*)d_in[16];
    const float* W_h1   = (const float*)d_in[17];
    const float* b_h1   = (const float*)d_in[18];
    const float* W_h2   = (const float*)d_in[19];
    const float* b_h2   = (const float*)d_in[20];
    const float* W_h3   = (const float*)d_in[21];
    const float* b_h3   = (const float*)d_in[22];
    float* out = (float*)d_out;

    // ---- workspace bump allocator (256B aligned) ----
    char* base = (char*)d_ws;
    size_t off = 0;
    auto alloc = [&](size_t bytes) -> char* {
        char* p = base + off;
        off += (bytes + 255) & ~(size_t)255;
        return p;
    };
    float*  h0    = (float*) alloc((size_t)NROWS * DMODEL * 4);
    float*  h1    = (float*) alloc((size_t)NROWS * DMODEL * 4);
    float*  xzb   = (float*) alloc((size_t)NROWS * 2 * ED * 4);
    float*  xab   = (float*) alloc((size_t)NROWS * ED * 4);
    float*  dbc   = (float*) alloc((size_t)NROWS * DBC_N * 4);
    float*  dtb   = (float*) alloc((size_t)NROWS * ED * 4);
    __bf16* unb   = (__bf16*)alloc((size_t)NROWS * DMODEL * 2);
    __bf16* xTb   = (__bf16*)alloc((size_t)NROWS * FEAT * 2);
    __bf16* xab16 = (__bf16*)alloc((size_t)NROWS * ED * 2);
    __bf16* yb16  = (__bf16*)alloc((size_t)NROWS * ED * 2);
    __bf16* hlast = (__bf16*)alloc((size_t)BATCH * DMODEL * 2);
    __bf16* fb16  = (__bf16*)alloc((size_t)BATCH * 1024 * 2);
    __bf16* g1b   = (__bf16*)alloc((size_t)BATCH * 512 * 2);
    __bf16* g2b   = (__bf16*)alloc((size_t)BATCH * 256 * 2);
    __bf16* wtb   = (__bf16*)alloc((size_t)DMODEL * 2 * ED * 2);  // 32MB scratch

    // ---- embedding ----
    transpose_x_kernel<<<(BATCH * SEQ * FEAT + 255) / 256, 256, 0, stream>>>(x, xTb);
    launch_castT(W_emb, wtb, FEAT, DMODEL, stream);
    launch_gemm(xTb, wtb, b_emb, nullptr, h0, 0, DMODEL,
                NROWS, DMODEL, FEAT, ACT_NONE, stream);

    float* hcur = h0;
    float* hnxt = h1;
    const int nBLE = (int)(((size_t)NROWS * ED + 255) / 256);

    for (int layer = 0; layer < NLAYERS; ++layer) {
        const float* nw  = norm_w + (size_t)layer * DMODEL;
        const float* wi  = W_in   + (size_t)layer * DMODEL * 2 * ED;
        const float* bi  = b_in   + (size_t)layer * 2 * ED;
        const float* cw  = conv_w + (size_t)layer * ED * KCONV;
        const float* cb  = conv_b + (size_t)layer * ED;
        const float* wx  = W_x    + (size_t)layer * ED * DBC_N;
        const float* wdt = W_dt   + (size_t)layer * DT_RANK * ED;
        const float* bdt = b_dt   + (size_t)layer * ED;
        const float* al  = A_log  + (size_t)layer * ED * NSTATE;
        const float* dp  = Dp     + (size_t)layer * ED;
        const float* wo  = W_out  + (size_t)layer * ED * DMODEL;
        const float* bo  = b_out  + (size_t)layer * DMODEL;

        rmsnorm_kernel<<<NROWS, 256, 0, stream>>>(hcur, nw, unb);

        // xz = un @ W_in + b_in          (1024 x 2048 x 8192, fast path)
        launch_castT(wi, wtb, DMODEL, 2 * ED, stream);
        launch_gemm(unb, wtb, bi, nullptr, xzb, 0, 2 * ED,
                    NROWS, 2 * ED, DMODEL, ACT_NONE, stream);

        conv_silu_kernel<<<nBLE, 256, 0, stream>>>(xzb, cw, cb, xab, xab16);

        // dbc = xa @ W_x                 (1024 x 4096 x 34, generic)
        launch_castT(wx, wtb, ED, DBC_N, stream);
        launch_gemm(xab16, wtb, nullptr, nullptr, dbc, 0, DBC_N,
                    NROWS, DBC_N, ED, ACT_NONE, stream);

        dt_kernel<<<nBLE, 256, 0, stream>>>(dbc, wdt, bdt, dtb);
        scan_kernel<<<(BATCH * ED) / 256, 256, 0, stream>>>(
            dtb, xab, dbc, xzb, al, dp, yb16);

        // hnxt = hcur + y @ W_out + b_out (1024 x 4096 x 2048, fast path)
        launch_castT(wo, wtb, ED, DMODEL, stream);
        launch_gemm(yb16, wtb, bo, hcur, hnxt, 0, DMODEL,
                    NROWS, DMODEL, ED, ACT_NONE, stream);

        float* t = hcur; hcur = hnxt; hnxt = t;
    }

    // ---- head ----
    last_rows_cast_kernel<<<(BATCH * DMODEL + 255) / 256, 256, 0, stream>>>(hcur, hlast);
    launch_castT(W_f, wtb, DMODEL, 1024, stream);
    launch_gemm(hlast, wtb, b_f, nullptr, fb16, 1, 1024,
                BATCH, 1024, DMODEL, ACT_NONE, stream);
    launch_castT(W_h1, wtb, 1024, 512, stream);
    launch_gemm(fb16, wtb, b_h1, nullptr, g1b, 1, 512,
                BATCH, 512, 1024, ACT_SILU, stream);
    launch_castT(W_h2, wtb, 512, 256, stream);
    launch_gemm(g1b, wtb, b_h2, nullptr, g2b, 1, 256,
                BATCH, 256, 512, ACT_LRELU, stream);
    head_out_kernel<<<1, 32, 0, stream>>>(g2b, W_h3, b_h3, out);
}
